// LayerwiseLowRankUplift_65773129171680
// MI455X (gfx1250) — matrix operand order repaired and checked by reading, not compiled, not run on hardware
//
#include <hip/hip_runtime.h>

// ---------------------------------------------------------------------------
// LayerwiseLowRankUplift for MI455X (gfx1250, wave32, WMMA bf16)
//   out[b,:] = z[b,:] + U[l] (V[l]^T z[b,:]),  l = layer_ids[b]
//
// Two-phase batched-GEMM formulation after binning samples by layer:
//   phase 1 (k_proj):  proj_tile(16x64) = Z_tile(16xH) * V_l(HxR)
//   phase 2 (k_apply): out_tile = proj_tile * U_l^T + z_tile
// proj tiles live in workspace global memory (<=768KB, L2-resident), so
// phase 2 can fan out to (tile x 16 column-slices) waves and saturate the
// 23.3 TB/s HBM on the bulk z-in/out traffic. bf16 WMMA, f32 accumulate.
// ---------------------------------------------------------------------------

typedef __bf16 v16bf __attribute__((ext_vector_type(16)));
typedef float  v8f   __attribute__((ext_vector_type(8)));
typedef float  v4f   __attribute__((ext_vector_type(4)));

#define NUM_LAYERS 32
#define HIDDEN     2048
#define RANK       64
#define BATCH      2048
#define TM         16      // samples per WMMA tile (M)
#define BK         32      // K per v_wmma_f32_16x16x32_bf16

// workspace layout (ints)
#define WS_COUNTS  0
#define WS_OFFS    32
#define WS_CURS    64
#define WS_NTILES  96
#define WS_IDX     128
#define WS_TILES   (128 + BATCH)
#define MAX_TILES  192                 // worst case sum ceil(c_l/16) = 158
#define WS_PROJ    2560                // int offset; float region, 16B aligned
// ws bytes needed: (2560 + MAX_TILES*TM*RANK) * 4  ~= 778 KB

#define WAVES_PER_BLOCK 8
#define GRID_PROJ   20                 // 160 wave slots >= 158 tiles
#define SLICES_PER_TILE 16             // phase-2 column slices per tile
#define HT_PER_SLICE (HIDDEN / 16 / SLICES_PER_TILE)   // 8 x 16 cols = 128
#define GRID_APPLY  320                // 2560 wave slots >= 158*16 = 2528

__device__ __forceinline__ __bf16 tob(float f) { return (__bf16)f; }

// ---- binning ---------------------------------------------------------------
__global__ void k_zero(int* ws) {
  int t = threadIdx.x;
  if (t < 32) { ws[WS_COUNTS + t] = 0; ws[WS_CURS + t] = 0; }
  if (t == 0) ws[WS_NTILES] = 0;
}

__global__ void k_count(const int* __restrict__ lid, int* ws) {
  int i = blockIdx.x * blockDim.x + threadIdx.x;
  if (i < BATCH) atomicAdd(&ws[WS_COUNTS + lid[i]], 1);
}

__global__ void k_scan(int* ws) {
  if (threadIdx.x == 0) {
    int off = 0, nt = 0;
    for (int l = 0; l < NUM_LAYERS; ++l) {
      int c = ws[WS_COUNTS + l];
      ws[WS_OFFS + l] = off;
      int tiles = (c + TM - 1) / TM;
      for (int t = 0; t < tiles; ++t) ws[WS_TILES + nt + t] = (l << 16) | t;
      nt += tiles;
      off += c;
    }
    ws[WS_NTILES] = nt;
  }
}

__global__ void k_scatter(const int* __restrict__ lid, int* ws) {
  int i = blockIdx.x * blockDim.x + threadIdx.x;
  if (i < BATCH) {
    int l = lid[i];
    int p = atomicAdd(&ws[WS_CURS + l], 1);
    ws[WS_IDX + ws[WS_OFFS + l] + p] = i;
  }
}

// ---- phase 1: proj(16x64) = Z_tile * V_l, one tile per wave32 --------------
__global__ void __launch_bounds__(WAVES_PER_BLOCK * 32)
k_proj(const float* __restrict__ z, const float* __restrict__ v,
       int* __restrict__ ws) {
  const int lane = threadIdx.x & 31;
  const int wave = threadIdx.x >> 5;
  const int tile = blockIdx.x * WAVES_PER_BLOCK + wave;
  if (tile >= ws[WS_NTILES]) return;

  const int hi = lane >> 4;    // lane half (ISA 7.12.2 layout split)
  const int lm = lane & 15;    // M (A rows) / N (B cols) within half

  const int enc   = ws[WS_TILES + tile];
  const int layer = enc >> 16;
  const int t     = enc & 0xffff;
  const int base  = ws[WS_OFFS + layer] + t * TM;
  const int cnt   = min(ws[WS_COUNTS + layer] - t * TM, TM);
  // A-matrix row for this lane; padded rows clamp to last valid sample
  // (duplicates compute identical values; phase-2 stores are guarded).
  const int srowA = ws[WS_IDX + base + min(lm, cnt - 1)];

  const float* vl = v + (size_t)layer * HIDDEN * RANK;

  v8f acc[4];                  // proj: 16 samples x 64 ranks (4 N-tiles)
  #pragma unroll
  for (int rt = 0; rt < 4; ++rt)
    #pragma unroll
    for (int j = 0; j < 8; ++j) acc[rt][j] = 0.0f;

  for (int kc = 0; kc < HIDDEN; kc += BK) {
    // A fragment (16x32 bf16): lane = row M=lm; elems 0..7 -> K=kc+hi*8+e,
    // elems 8..15 -> K=kc+16+hi*8+(e-8)           (ISA 7.12.2, 16-bit A)
    const float* zr = z + (size_t)srowA * HIDDEN + kc + hi * 8;
    v4f za0 = *(const v4f*)(zr);
    v4f za1 = *(const v4f*)(zr + 4);
    v4f za2 = *(const v4f*)(zr + 16);
    v4f za3 = *(const v4f*)(zr + 20);
    __builtin_prefetch(zr + BK, 0, 0);             // global_prefetch_b8
    v16bf a;
    #pragma unroll
    for (int e = 0; e < 4; ++e) {
      a[e]      = tob(za0[e]);
      a[e + 4]  = tob(za1[e]);
      a[e + 8]  = tob(za2[e]);
      a[e + 12] = tob(za3[e]);
    }
    // B fragments (32x16 bf16): lane = column N=lm; elem e -> K=kc+hi*8+(e<8?e:e+8).
    // Across the wave each elem-e load covers two 64B contiguous segments.
    const float* vb = vl + (size_t)(kc + hi * 8) * RANK + lm;
    v16bf b0, b1, b2, b3;
    #pragma unroll
    for (int e = 0; e < 16; ++e) {
      const float* p = vb + (size_t)(e < 8 ? e : e + 8) * RANK;
      b0[e] = tob(p[0]);
      b1[e] = tob(p[16]);
      b2[e] = tob(p[32]);
      b3[e] = tob(p[48]);
    }
    // Same A operand for all four: set VOP3P reuse-A hint on instrs 2..4
    // (legal when current instruction is identical to the previous one).
    acc[0] = __builtin_amdgcn_wmma_f32_16x16x32_bf16(
        false, a, false, b0, (short)0, acc[0], false, false);
    acc[1] = __builtin_amdgcn_wmma_f32_16x16x32_bf16(
        false, a, false, b1, (short)0, acc[1], true, false);
    acc[2] = __builtin_amdgcn_wmma_f32_16x16x32_bf16(
        false, a, false, b2, (short)0, acc[2], true, false);
    acc[3] = __builtin_amdgcn_wmma_f32_16x16x32_bf16(
        false, a, false, b3, (short)0, acc[3], true, false);
  }

  // Park proj in workspace (L2-resident). D layout: VGPR j holds
  // M = j + hi*8, N = lm  ->  coalesced 2x64B segments per store.
  float* wp = (float*)(ws + WS_PROJ) + (size_t)tile * (TM * RANK);
  #pragma unroll
  for (int rt = 0; rt < 4; ++rt)
    #pragma unroll
    for (int j = 0; j < 8; ++j)
      wp[(j + hi * 8) * RANK + rt * 16 + lm] = acc[rt][j];
}

// ---- phase 2: out = proj * U_l^T + z, (tile x slice) per wave32 ------------
__global__ void __launch_bounds__(WAVES_PER_BLOCK * 32)
k_apply(const float* __restrict__ z, const float* __restrict__ u,
        const int* __restrict__ ws, float* __restrict__ out) {
  const int lane = threadIdx.x & 31;
  const int wave = threadIdx.x >> 5;
  const int sid  = blockIdx.x * WAVES_PER_BLOCK + wave;
  const int tile = sid / SLICES_PER_TILE;
  const int slice = sid - tile * SLICES_PER_TILE;
  if (tile >= ws[WS_NTILES]) return;

  const int hi = lane >> 4;
  const int lm = lane & 15;

  const int enc   = ws[WS_TILES + tile];
  const int layer = enc >> 16;
  const int t     = enc & 0xffff;
  const int base  = ws[WS_OFFS + layer] + t * TM;
  const int cnt   = min(ws[WS_COUNTS + layer] - t * TM, TM);

  const float* ul = u + (size_t)layer * HIDDEN * RANK;

  // C/D-layout rows for this lane: M = j + hi*8
  int  srows[8];
  bool valid[8];
  #pragma unroll
  for (int j = 0; j < 8; ++j) {
    const int m = j + hi * 8;
    valid[j] = (m < cnt);
    srows[j] = ws[WS_IDX + base + min(m, cnt - 1)];
  }

  // A2 fragments (proj as bf16) from workspace, hoisted over the h loop.
  // K=64 split into two K=32 WMMAs (a2lo: r in [0,32), a2hi: [32,64)).
  v16bf a2lo, a2hi;
  const float* pr =
      (const float*)(ws + WS_PROJ) + (size_t)tile * (TM * RANK) + lm * RANK;
  #pragma unroll
  for (int e = 0; e < 16; ++e) {
    const int r = e + hi * 16;
    a2lo[e] = tob(pr[r]);
    a2hi[e] = tob(pr[r + 32]);
  }

  const int ht0 = slice * HT_PER_SLICE;
  for (int ht = ht0; ht < ht0 + HT_PER_SLICE; ++ht) {
    const int hbase = ht * 16;

    // Seed accumulator with z rows: D = A*B + C gives z + delta for free.
    v8f c;
    #pragma unroll
    for (int j = 0; j < 8; ++j)
      c[j] = z[(size_t)srows[j] * HIDDEN + hbase + lm];

    // B2 fragment: B[K=r][N=h] = U_l[hbase+lm][r]; elem e -> r = e + hi*16
    const float* ub = ul + (size_t)(hbase + lm) * RANK + hi * 16;
    v4f f0 = *(const v4f*)(ub);
    v4f f1 = *(const v4f*)(ub + 4);
    v4f f2 = *(const v4f*)(ub + 8);
    v4f f3 = *(const v4f*)(ub + 12);
    v4f g0 = *(const v4f*)(ub + 32);
    v4f g1 = *(const v4f*)(ub + 36);
    v4f g2 = *(const v4f*)(ub + 40);
    v4f g3 = *(const v4f*)(ub + 44);
    v16bf b2lo, b2hi;
    #pragma unroll
    for (int e = 0; e < 4; ++e) {
      b2lo[e]      = tob(f0[e]);
      b2lo[e + 4]  = tob(f1[e]);
      b2lo[e + 8]  = tob(f2[e]);
      b2lo[e + 12] = tob(f3[e]);
      b2hi[e]      = tob(g0[e]);
      b2hi[e + 4]  = tob(g1[e]);
      b2hi[e + 8]  = tob(g2[e]);
      b2hi[e + 12] = tob(g3[e]);
    }

    c = __builtin_amdgcn_wmma_f32_16x16x32_bf16(
        false, a2lo, false, b2lo, (short)0, c, false, false);
    c = __builtin_amdgcn_wmma_f32_16x16x32_bf16(
        false, a2hi, false, b2hi, (short)0, c, false, false);

    #pragma unroll
    for (int j = 0; j < 8; ++j)
      if (valid[j])
        out[(size_t)srows[j] * HIDDEN + hbase + lm] = c[j];
  }
}

// ---------------------------------------------------------------------------
extern "C" void kernel_launch(void* const* d_in, const int* in_sizes, int n_in,
                              void* d_out, int out_size, void* d_ws, size_t ws_size,
                              hipStream_t stream) {
  const float* z   = (const float*)d_in[0];
  const int*   lid = (const int*)d_in[1];
  const float* u   = (const float*)d_in[2];
  const float* v   = (const float*)d_in[3];
  float* out = (float*)d_out;
  int*   ws  = (int*)d_ws;   // needs (2560 + 192*1024)*4 ~= 778 KB

  k_zero   <<<1, 64, 0, stream>>>(ws);
  k_count  <<<BATCH / 256, 256, 0, stream>>>(lid, ws);
  k_scan   <<<1, 32, 0, stream>>>(ws);
  k_scatter<<<BATCH / 256, 256, 0, stream>>>(lid, ws);
  k_proj   <<<GRID_PROJ,  WAVES_PER_BLOCK * 32, 0, stream>>>(z, v, ws);
  k_apply  <<<GRID_APPLY, WAVES_PER_BLOCK * 32, 0, stream>>>(z, u, ws, out);
}
